// MultiHeadAttentionn_52432960749570
// MI455X (gfx1250) — matrix operand order repaired and checked by reading, not compiled, not run on hardware
//
#include <hip/hip_runtime.h>
#include <hip/hip_bf16.h>

typedef __attribute__((ext_vector_type(8)))  __bf16 bf16x8;
typedef __attribute__((ext_vector_type(16))) __bf16 bf16x16;
typedef __attribute__((ext_vector_type(8)))  float  f32x8;
typedef __attribute__((ext_vector_type(4)))  unsigned u32x4;
typedef __attribute__((ext_vector_type(8)))  unsigned u32x8;
typedef unsigned short ushort_t;

// ---------- helpers ----------

__device__ __forceinline__ ushort_t f2bf(float f) {
    unsigned u = __builtin_bit_cast(unsigned, f);
    unsigned r = u + 0x7FFFu + ((u >> 16) & 1u);   // round-to-nearest-even
    return (ushort_t)(r >> 16);
}

__device__ __forceinline__ bf16x16 cat8(bf16x8 lo, bf16x8 hi) {
    return __builtin_shufflevector(lo, hi, 0,1,2,3,4,5,6,7,8,9,10,11,12,13,14,15);
}

// A fragment (16x32 bf16, MxK): lanes 0-15 hold K 0..7 & 16..23, lanes 16-31 hold 8..15 & 24..31.
__device__ __forceinline__ bf16x16 load_afrag(const ushort_t* p0, int pitch, int lane) {
    const ushort_t* p = p0 + (lane & 15) * pitch + ((lane < 16) ? 0 : 8);
    bf16x8 lo = *(const bf16x8*)(p);
    bf16x8 hi = *(const bf16x8*)(p + 16);
    return cat8(lo, hi);
}

// B fragment (32x16 bf16, KxN): lane holds column N=lane%16; lanes 0-15 hold K 0..7 & 16..23, lanes 16-31 hold 8..15 & 24..31.
// Source tile stored [N][K] (K-contiguous) so each lane reads two 8-element bf16 runs.
__device__ __forceinline__ bf16x16 load_bfrag(const ushort_t* p0, int pitch, int lane) {
    const ushort_t* p = p0 + (lane & 15) * pitch + ((lane < 16) ? 0 : 8);
    bf16x8 lo = *(const bf16x8*)(p);
    bf16x8 hi = *(const bf16x8*)(p + 16);
    return cat8(lo, hi);
}

__device__ __forceinline__ f32x8 wmma_bf16(bf16x16 a, bf16x16 b, f32x8 c) {
    return __builtin_amdgcn_wmma_f32_16x16x32_bf16(false, a, false, b, (short)0, c, false, false);
}

// ---------- Tensor Data Mover (TDM) descriptor helpers ----------
// D# per CDNA5 ISA. 2D tile, bf16 elements (data_size=1).
// Optional LDS padding: pad_iv=4 -> every 32 DWORDs (one 64-elem bf16 row),
// pad_amt=3 -> insert 4 DWORDs (8 bf16) => padded LDS pitch of 72 elements.

struct TdmDesc { u32x4 g0; u32x8 g1; };

__device__ __forceinline__ TdmDesc tdm_make(unsigned lds_addr,
                                            unsigned long long gaddr,
                                            unsigned tile_d0, unsigned tile_d1,
                                            unsigned tensor_d0, unsigned tensor_d1,
                                            unsigned long long stride_d0,
                                            bool pad_en, unsigned pad_iv, unsigned pad_amt)
{
    TdmDesc d;
    // group0: count=1 (valid), lds_addr, 57-bit global byte address, type=2 ("image")
    d.g0[0] = 1u;
    d.g0[1] = lds_addr;
    d.g0[2] = (unsigned)gaddr;
    d.g0[3] = ((unsigned)(gaddr >> 32) & 0x01FFFFFFu) | (2u << 30);
    // group1
    unsigned w0 = 1u << 16;                         // data_size = 1 (2-byte elements)
    if (pad_en) w0 |= (1u << 20) | (pad_iv << 22) | (pad_amt << 25);
    d.g1[0] = w0;                                   // workgroup_mask=0 (not in cluster)
    d.g1[1] = (tensor_d0 & 0xFFFFu) << 16;          // tensor_dim0[15:0] @ bits[63:48]
    d.g1[2] = (tensor_d0 >> 16) | ((tensor_d1 & 0xFFFFu) << 16);
    d.g1[3] = (tensor_d1 >> 16) | ((tile_d0 & 0xFFFFu) << 16);
    d.g1[4] = tile_d1 & 0xFFFFu;                    // tile_dim1; tile_dim2=0 (2D)
    d.g1[5] = (unsigned)stride_d0;                  // tensor_dim0_stride[31:0]
    d.g1[6] = (unsigned)(stride_d0 >> 32) & 0xFFFFu;// stride[47:32]; dim1_stride=0
    d.g1[7] = 0u;
    return d;
}

__device__ __forceinline__ void tdm_load_to_lds(const TdmDesc& d) {
    u32x4 z = (u32x4)(0u);
    asm volatile("tensor_load_to_lds %0, %1, %2, %3"
                 :: "s"(d.g0), "s"(d.g1), "s"(z), "s"(z) : "memory");
}

__device__ __forceinline__ void tdm_store_from_lds(const TdmDesc& d) {
    u32x4 z = (u32x4)(0u);
    asm volatile("tensor_store_from_lds %0, %1, %2, %3"
                 :: "s"(d.g0), "s"(d.g1), "s"(z), "s"(z) : "memory");
}

__device__ __forceinline__ unsigned lds_off(const void* p) {
    return (unsigned)(unsigned long long)p;   // generic LDS addr: low 32 bits = LDS byte offset
}

// ---------- GEMM: out[m,n] = sum_k A[m,k] * W[n,k] + bias[n] ----------
// A: MxK row-major (f32: VGPR-staged+converted, bf16: TDM async tile load);
// W: NxK row-major f32; out: MxN row-major (bf16 or f32).
// Tile 128x128, BK=64, 256 threads (8 waves, 2x4), each wave 64x32 (4x2 WMMA tiles).

#define TM 128
#define TN 128
#define BK 64
#define LP 72   // LDS pitch in bf16 elements (64 + 8 pad -> conflict-free ds_load_b128)

template<bool A_IS_F32, bool OUT_IS_F32>
__global__ __launch_bounds__(256)
void gemm_nt_bf16(const void* __restrict__ Aptr,
                  const float* __restrict__ Wptr,
                  const float* __restrict__ bias,
                  void* __restrict__ Optr,
                  int M, int N, int K)
{
    __shared__ ushort_t sA[TM][LP];
    __shared__ ushort_t sB[TN][LP];

    const int tid  = threadIdx.x;
    const int lane = tid & 31;
    const int wave = tid >> 5;
    const int wm   = (wave >> 2) * 64;   // wave M offset: 0 / 64
    const int wn   = (wave & 3) * 32;    // wave N offset: 0/32/64/96

    const int mBase = blockIdx.y * TM;
    const int nBase = blockIdx.x * TN;

    const int lr = tid >> 4;          // 0..15
    const int lc = (tid & 15) * 4;    // 0..60  (4-element column group)

    f32x8 acc[4][2];
    #pragma unroll
    for (int mi = 0; mi < 4; ++mi)
        #pragma unroll
        for (int ni = 0; ni < 2; ++ni)
            acc[mi][ni] = (f32x8)(0.0f);

    for (int k0 = 0; k0 < K; k0 += BK) {
        // ---- stage global -> regs ----
        float4 av[8]; float4 bv[8];
        #pragma unroll
        for (int i = 0; i < 8; ++i) {
            const int r = lr + i * 16;
            if constexpr (A_IS_F32) {
                av[i] = *(const float4*)((const float*)Aptr + (size_t)(mBase + r) * K + k0 + lc);
            }
            bv[i] = *(const float4*)(Wptr + (size_t)(nBase + r) * K + k0 + lc);
        }
        __syncthreads();   // previous compute finished reading LDS
        // ---- A tile: TDM async copy (bf16 source), or converted VGPR staging (f32) ----
        if constexpr (!A_IS_F32) {
            if (wave == 0) {
                const unsigned long long ga =
                    (unsigned long long)((const ushort_t*)Aptr + (size_t)mBase * K + k0);
                tdm_load_to_lds(tdm_make(lds_off(&sA[0][0]), ga,
                                         /*tile*/ BK, TM, /*tensor*/ 4096u, 4096u,
                                         /*stride*/ (unsigned long long)K,
                                         true, 4u, 3u));
            }
        }
        // ---- regs -> LDS (convert f32 -> bf16) ----
        #pragma unroll
        for (int i = 0; i < 8; ++i) {
            const int r = lr + i * 16;
            if constexpr (A_IS_F32) {
                uint2 p;
                p.x = (unsigned)f2bf(av[i].x) | ((unsigned)f2bf(av[i].y) << 16);
                p.y = (unsigned)f2bf(av[i].z) | ((unsigned)f2bf(av[i].w) << 16);
                *(uint2*)&sA[r][lc] = p;
            }
            uint2 q;
            q.x = (unsigned)f2bf(bv[i].x) | ((unsigned)f2bf(bv[i].y) << 16);
            q.y = (unsigned)f2bf(bv[i].z) | ((unsigned)f2bf(bv[i].w) << 16);
            *(uint2*)&sB[r][lc] = q;
        }
        if constexpr (!A_IS_F32) {
            if (wave == 0) __builtin_amdgcn_s_wait_tensorcnt(0);
        }
        __syncthreads();
        // ---- compute: 2 K-steps of 32 ----
        #pragma unroll
        for (int ks = 0; ks < BK; ks += 32) {
            bf16x16 af[4], bfv[2];
            #pragma unroll
            for (int mi = 0; mi < 4; ++mi)
                af[mi] = load_afrag(&sA[wm + mi * 16][ks], LP, lane);
            #pragma unroll
            for (int ni = 0; ni < 2; ++ni)
                bfv[ni] = load_bfrag(&sB[wn + ni * 16][ks], LP, lane);
            #pragma unroll
            for (int mi = 0; mi < 4; ++mi)
                #pragma unroll
                for (int ni = 0; ni < 2; ++ni)
                    acc[mi][ni] = wmma_bf16(af[mi], bfv[ni], acc[mi][ni]);
        }
    }

    // ---- epilogue: D layout VGPR r -> M = r (lanes 0-15) / r+8 (lanes 16-31), N = lane%16 ----
    const int nl   = lane & 15;
    const int rofs = (lane < 16) ? 0 : 8;
    #pragma unroll
    for (int ni = 0; ni < 2; ++ni) {
        const int n = nBase + wn + ni * 16 + nl;
        const float bb = bias[n];
        #pragma unroll
        for (int mi = 0; mi < 4; ++mi) {
            #pragma unroll
            for (int r = 0; r < 8; ++r) {
                const int m = mBase + wm + mi * 16 + r + rofs;
                const float v = acc[mi][ni][r] + bb;
                if constexpr (OUT_IS_F32)
                    ((float*)Optr)[(size_t)m * N + n] = v;
                else
                    ((ushort_t*)Optr)[(size_t)m * N + n] = f2bf(v);
            }
        }
    }
}

// ---------- per-token attention over heads ----------
// One 128-thread block (4 waves) per token (b,s).
// TDM loads q/k/v tiles into padded LDS; scores via WMMA; row softmax;
// vals via WMMA; TDM stores the merged-transposed tile (stride 4096).

__global__ __launch_bounds__(128)
void attn_heads_kernel(const ushort_t* __restrict__ Q,
                       const ushort_t* __restrict__ Km,
                       const ushort_t* __restrict__ V,
                       ushort_t* __restrict__ Mg)
{
    __shared__ ushort_t sQ[64][LP];
    __shared__ ushort_t sK[64][LP];
    __shared__ ushort_t sV[64][LP];
    __shared__ ushort_t sP[64][LP];
    __shared__ float    sS[64][65];
    __shared__ ushort_t sO[64][64];    // unpadded: TDM store does no de-padding

    const int tok = blockIdx.x;          // b*64 + s
    const int b = tok >> 6, s = tok & 63;
    const size_t base = (size_t)tok * 4096;
    const int tid  = threadIdx.x;
    const int lane = tid & 31;
    const int wave = tid >> 5;
    const int mrow = wave * 16;          // each wave owns a 16-row band

    // ---- async TDM stage q,k,v (64x64 bf16 each) into padded LDS ----
    if (wave == 0) {
        tdm_load_to_lds(tdm_make(lds_off(&sQ[0][0]), (unsigned long long)(Q + base),
                                 64u, 64u, 4096u, 4096u, 64ull, true, 4u, 3u));
        tdm_load_to_lds(tdm_make(lds_off(&sK[0][0]), (unsigned long long)(Km + base),
                                 64u, 64u, 4096u, 4096u, 64ull, true, 4u, 3u));
        tdm_load_to_lds(tdm_make(lds_off(&sV[0][0]), (unsigned long long)(V + base),
                                 64u, 64u, 4096u, 4096u, 64ull, true, 4u, 3u));
        __builtin_amdgcn_s_wait_tensorcnt(0);
    }
    __syncthreads();

    const int nl   = lane & 15;
    const int rofs = (lane < 16) ? 0 : 8;

    // scores = q @ k^T, scaled by 1/sqrt(64)
    #pragma unroll
    for (int ni = 0; ni < 4; ++ni) {
        f32x8 acc = (f32x8)(0.0f);
        #pragma unroll
        for (int ks = 0; ks < 64; ks += 32) {
            bf16x16 a = load_afrag(&sQ[mrow][ks],     LP, lane);
            bf16x16 bb= load_bfrag(&sK[ni * 16][ks],  LP, lane);
            acc = wmma_bf16(a, bb, acc);
        }
        #pragma unroll
        for (int r = 0; r < 8; ++r)
            sS[mrow + r + rofs][ni * 16 + nl] = acc[r] * 0.125f;
    }
    __syncthreads();

    // row softmax (64 rows, one thread each)
    if (tid < 64) {
        float mx = -3.4e38f;
        #pragma unroll 8
        for (int j = 0; j < 64; ++j) mx = fmaxf(mx, sS[tid][j]);
        float sum = 0.0f;
        #pragma unroll 8
        for (int j = 0; j < 64; ++j) sum += __expf(sS[tid][j] - mx);
        const float inv = 1.0f / sum;
        #pragma unroll 8
        for (int j = 0; j < 64; ++j)
            sP[tid][j] = f2bf(__expf(sS[tid][j] - mx) * inv);
    }
    __syncthreads();

    // vals = attn @ v^T
    #pragma unroll
    for (int ni = 0; ni < 4; ++ni) {
        f32x8 acc = (f32x8)(0.0f);
        #pragma unroll
        for (int ks = 0; ks < 64; ks += 32) {
            bf16x16 a = load_afrag(&sP[mrow][ks],    LP, lane);
            bf16x16 bb= load_bfrag(&sV[ni * 16][ks], LP, lane);
            acc = wmma_bf16(a, bb, acc);
        }
        #pragma unroll
        for (int r = 0; r < 8; ++r)
            sO[mrow + r + rofs][ni * 16 + nl] = f2bf(acc[r]);
    }
    __syncthreads();

    // ---- TDM store: merged[b][i][s*64 + j] (transpose folded into stride-4096 tile store) ----
    if (wave == 0) {
        const size_t ob = (size_t)b * 262144 + (size_t)s * 64;
        tdm_store_from_lds(tdm_make(lds_off(&sO[0][0]), (unsigned long long)(Mg + ob),
                                    64u, 64u, 4096u, 4096u, 4096ull, false, 0u, 0u));
        __builtin_amdgcn_s_wait_tensorcnt(0);
    }
}

// ---------- launch ----------

extern "C" void kernel_launch(void* const* d_in, const int* in_sizes, int n_in,
                              void* d_out, int out_size, void* d_ws, size_t ws_size,
                              hipStream_t stream)
{
    (void)in_sizes; (void)n_in; (void)out_size; (void)ws_size;
    const float* x  = (const float*)d_in[0];
    const float* Wq = (const float*)d_in[1];
    const float* bq = (const float*)d_in[2];
    const float* Wk = (const float*)d_in[3];
    const float* bk = (const float*)d_in[4];
    const float* Wv = (const float*)d_in[5];
    const float* bv = (const float*)d_in[6];
    const float* Wp = (const float*)d_in[7];
    const float* bp = (const float*)d_in[8];

    const size_t MAT = 4096ull * 4096ull;     // elements per bf16 matrix
    ushort_t* qb = (ushort_t*)d_ws;
    ushort_t* kb = qb + MAT;
    ushort_t* vb = kb + MAT;
    ushort_t* mb = vb + MAT;

    dim3 grid(32, 32), blk(256);
    gemm_nt_bf16<true,  false><<<grid, blk, 0, stream>>>(x,  Wq, bq, qb, 4096, 4096, 4096);
    gemm_nt_bf16<true,  false><<<grid, blk, 0, stream>>>(x,  Wk, bk, kb, 4096, 4096, 4096);
    gemm_nt_bf16<true,  false><<<grid, blk, 0, stream>>>(x,  Wv, bv, vb, 4096, 4096, 4096);

    attn_heads_kernel<<<dim3(4096), dim3(128), 0, stream>>>(qb, kb, vb, mb);

    gemm_nt_bf16<false, true ><<<grid, blk, 0, stream>>>(mb, Wp, bp, (float*)d_out, 4096, 4096, 4096);
}
